// WoodStressGNN_31585189495032
// MI455X (gfx1250) — matrix operand order repaired and checked by reading, not compiled, not run on hardware
//
#include <hip/hip_runtime.h>
#include <hip/hip_bf16.h>

#define N_NODES 100000
#define N_EDGES 1600000
#define HID 64
#define N_TILES (N_NODES / 16)   // 6250, exact

typedef __attribute__((ext_vector_type(16))) _Float16 v16h;
typedef __attribute__((ext_vector_type(8)))  float    v8f;

union frag16 { v16h v; _Float16 h[16]; };

// ---------------------------------------------------------------- utility
__global__ void zero_f32_kernel(float* __restrict__ p, int n) {
    int i = blockIdx.x * blockDim.x + threadIdx.x;
    if (i < n) p[i] = 0.0f;
}

// degree: one thread per edge
__global__ void deg_kernel(const int* __restrict__ dst, float* __restrict__ deg, int n_edges) {
    int e = blockIdx.x * blockDim.x + threadIdx.x;
    if (e < n_edges) atomicAdd(&deg[dst[e]], 1.0f);
}

// convert + zero-pad all six weight matrices to f16 [64][64], out-major
__global__ void prep_w_kernel(const float* __restrict__ w1l, const float* __restrict__ w1r,
                              const float* __restrict__ w2l, const float* __restrict__ w2r,
                              const float* __restrict__ w3l, const float* __restrict__ w3r,
                              _Float16* __restrict__ w16) {
    int m = blockIdx.x;            // 0..5
    const float* src; int ic;
    switch (m) {
        case 0: src = w1l; ic = 6;  break;
        case 1: src = w1r; ic = 6;  break;
        case 2: src = w2l; ic = 64; break;
        case 3: src = w2r; ic = 64; break;
        case 4: src = w3l; ic = 64; break;
        default: src = w3r; ic = 64; break;
    }
    for (int i = threadIdx.x; i < 64 * 64; i += blockDim.x) {
        int o = i >> 6, k = i & 63;
        float v = (k < ic) ? src[o * ic + k] : 0.0f;
        w16[m * 4096 + i] = (_Float16)v;
    }
}

// pad input x [N,6] f32 -> [N,64] f16 (zeros beyond ch 6)
__global__ void pad_x_kernel(const float* __restrict__ x, _Float16* __restrict__ h, int total) {
    int i = blockIdx.x * blockDim.x + threadIdx.x;
    if (i >= total) return;
    int node = i >> 6, c = i & 63;
    h[i] = (c < 6) ? (_Float16)x[node * 6 + c] : (_Float16)0.0f;
}

// layer-1 scatter: 8 threads/edge, 6 valid channels, fp32 gather + fp32 atomics
__global__ void scatter6_kernel(const float* __restrict__ x, const int* __restrict__ src,
                                const int* __restrict__ dst, float* __restrict__ agg, int n_edges) {
    int t = blockIdx.x * blockDim.x + threadIdx.x;
    int e = t >> 3, c = t & 7;
    if (e >= n_edges || c >= 6) return;
    float v = x[src[e] * 6 + c];
    atomicAdd(&agg[dst[e] * HID + c], v);
}

// layers 2/3 scatter: 16 threads/edge, 4 channels each; f16 gather, fp32 atomic accumulate
__global__ void scatter_h_kernel(const _Float16* __restrict__ h, const int* __restrict__ src,
                                 const int* __restrict__ dst, float* __restrict__ agg, int n_edges) {
    int t = blockIdx.x * blockDim.x + threadIdx.x;
    int e = t >> 4;
    if (e >= n_edges) return;
    int c4 = (t & 15) * 4;
    int s = src[e], d = dst[e];
    const _Float16* p = h + (size_t)s * HID + c4;
    float* q = agg + (size_t)d * HID + c4;
    atomicAdd(q + 0, (float)p[0]);
    atomicAdd(q + 1, (float)p[1]);
    atomicAdd(q + 2, (float)p[2]);
    atomicAdd(q + 3, (float)p[3]);
}

// ---------------------------------------------------------------- fused SAGE layer (WMMA)
// hout[16x64 tile] = relu( (agg/deg) @ wl^T + hin @ wr^T + b ), all per 16-node tile.
__global__ void __launch_bounds__(256)
sage_wmma_kernel(const float* __restrict__ agg, const float* __restrict__ deg,
                 const _Float16* __restrict__ hin,
                 const _Float16* __restrict__ wl,   // [64][64] f16, out-major
                 const _Float16* __restrict__ wr,   // [64][64] f16, out-major
                 const float* __restrict__ bias,    // [64]
                 _Float16* __restrict__ hout,
                 int n_tiles) {
    const int lane   = threadIdx.x & 31;
    const int col    = lane & 15;          // A-row / B-col / C-col index
    const int hi     = lane >> 4;          // half-wave select
    const int wid    = (blockIdx.x * blockDim.x + threadIdx.x) >> 5;
    const int nwaves = (gridDim.x * blockDim.x) >> 5;

    // B fragments (weights) live in registers for the whole tile loop.
    // B k order used here (same as A): lane<16 -> col=lane, K=0..15; lane>=16 -> col=lane-16, K=16..31.
    v16h Bl[4][2], Br[4][2];
#pragma unroll
    for (int nt = 0; nt < 4; ++nt)
#pragma unroll
        for (int kc = 0; kc < 2; ++kc) {
            int off = (nt * 16 + col) * HID + kc * 32 + hi * 16;  // 32B aligned
            Bl[nt][kc] = *(const v16h*)(wl + off);
            Br[nt][kc] = *(const v16h*)(wr + off);
        }

    for (int tile = wid; tile < n_tiles; tile += nwaves) {
        const int rowA = tile * 16 + col;          // node row this lane supplies to A
        const float inv = 1.0f / fmaxf(deg[rowA], 1.0f);
        const int kofs = hi * 16;

        // A k order used here (same as B): halfs 0..15 -> K=kofs..kofs+15
        frag16 Aagg[2], Aself[2];
#pragma unroll
        for (int kc = 0; kc < 2; ++kc) {
            const int base = rowA * HID + kc * 32 + kofs;
#pragma unroll
            for (int g = 0; g < 2; ++g) {
                const float*    pa = agg + base + g * 8;
                const _Float16* ph = hin + base + g * 8;
#pragma unroll
                for (int j = 0; j < 8; ++j) {
                    Aagg[kc].h[g * 8 + j]  = (_Float16)(pa[j] * inv);
                    Aself[kc].h[g * 8 + j] = ph[j];
                }
            }
        }

#pragma unroll
        for (int nt = 0; nt < 4; ++nt) {
            v8f c = {};
            c = __builtin_amdgcn_wmma_f32_16x16x32_f16(false, Aagg[0].v,  false, Bl[nt][0], (short)0, c, false, false);
            c = __builtin_amdgcn_wmma_f32_16x16x32_f16(false, Aagg[1].v,  false, Bl[nt][1], (short)0, c, false, false);
            c = __builtin_amdgcn_wmma_f32_16x16x32_f16(false, Aself[0].v, false, Br[nt][0], (short)0, c, false, false);
            c = __builtin_amdgcn_wmma_f32_16x16x32_f16(false, Aself[1].v, false, Br[nt][1], (short)0, c, false, false);

            // C 16x16 layout: VGPR r -> row r + hi*8, col = lane&15
            const int outcol = nt * 16 + col;
            const float b = bias[outcol];
#pragma unroll
            for (int r = 0; r < 8; ++r) {
                const int outrow = tile * 16 + r + hi * 8;
                float v = c[r] + b;
                v = v > 0.0f ? v : 0.0f;
                hout[(size_t)outrow * HID + outcol] = (_Float16)v;
            }
        }
    }
}

// final FC: 64 -> 3, fp32 out
__global__ void fc_kernel(const _Float16* __restrict__ h, const float* __restrict__ fw,
                          const float* __restrict__ fb, float* __restrict__ out, int n) {
    int node = blockIdx.x * blockDim.x + threadIdx.x;
    if (node >= n) return;
    float a0 = fb[0], a1 = fb[1], a2 = fb[2];
    const _Float16* p = h + (size_t)node * HID;
#pragma unroll 8
    for (int k = 0; k < HID; ++k) {
        float v = (float)p[k];
        a0 += v * fw[0 * HID + k];
        a1 += v * fw[1 * HID + k];
        a2 += v * fw[2 * HID + k];
    }
    out[node * 3 + 0] = a0;
    out[node * 3 + 1] = a1;
    out[node * 3 + 2] = a2;
}

// ---------------------------------------------------------------- launch
extern "C" void kernel_launch(void* const* d_in, const int* in_sizes, int n_in,
                              void* d_out, int out_size, void* d_ws, size_t ws_size,
                              hipStream_t stream) {
    const float* x   = (const float*)d_in[0];
    const int*   ei  = (const int*)d_in[1];
    const int*   src = ei;
    const int*   dst = ei + N_EDGES;
    const float* w1l = (const float*)d_in[2];
    const float* b1  = (const float*)d_in[3];
    const float* w1r = (const float*)d_in[4];
    const float* w2l = (const float*)d_in[5];
    const float* b2  = (const float*)d_in[6];
    const float* w2r = (const float*)d_in[7];
    const float* w3l = (const float*)d_in[8];
    const float* b3  = (const float*)d_in[9];
    const float* w3r = (const float*)d_in[10];
    const float* fw  = (const float*)d_in[11];
    const float* fb  = (const float*)d_in[12];
    float* out = (float*)d_out;

    // workspace carve-up (256B aligned)
    char* ws = (char*)d_ws;
    constexpr size_t OFF_DEG = 0;                                       // N f32
    constexpr size_t OFF_AGG = 400128;                                  // N*64 f32
    constexpr size_t OFF_HA  = OFF_AGG + (size_t)N_NODES * HID * 4;     // N*64 f16
    constexpr size_t OFF_HB  = OFF_HA + (size_t)N_NODES * HID * 2;      // N*64 f16
    constexpr size_t OFF_W16 = OFF_HB + (size_t)N_NODES * HID * 2;      // 6*4096 f16
    float*    deg = (float*)(ws + OFF_DEG);
    float*    agg = (float*)(ws + OFF_AGG);
    _Float16* hA  = (_Float16*)(ws + OFF_HA);
    _Float16* hB  = (_Float16*)(ws + OFF_HB);
    _Float16* w16 = (_Float16*)(ws + OFF_W16);

    const int TPB = 256;
    const int gZeroDeg = (N_NODES + TPB - 1) / TPB;
    const int gDeg     = (N_EDGES + TPB - 1) / TPB;
    const int gPad     = (N_NODES * HID + TPB - 1) / TPB;
    const int gZeroAgg = gPad;
    const int gSc6     = (N_EDGES * 8 + TPB - 1) / TPB;
    const int gSc16    = (N_EDGES * 16) / TPB;
    const int gGemm    = 208;   // 1664 waves over 6250 tiles
    const int gFc      = (N_NODES + TPB - 1) / TPB;

    // one-time prep
    zero_f32_kernel<<<gZeroDeg, TPB, 0, stream>>>(deg, N_NODES);
    deg_kernel<<<gDeg, TPB, 0, stream>>>(dst, deg, N_EDGES);
    prep_w_kernel<<<6, TPB, 0, stream>>>(w1l, w1r, w2l, w2r, w3l, w3r, w16);
    pad_x_kernel<<<gPad, TPB, 0, stream>>>(x, hA, N_NODES * HID);

    // layer 1
    zero_f32_kernel<<<gZeroAgg, TPB, 0, stream>>>(agg, N_NODES * HID);
    scatter6_kernel<<<gSc6, TPB, 0, stream>>>(x, src, dst, agg, N_EDGES);
    sage_wmma_kernel<<<gGemm, TPB, 0, stream>>>(agg, deg, hA, w16 + 0 * 4096, w16 + 1 * 4096, b1, hB, N_TILES);

    // layer 2
    zero_f32_kernel<<<gZeroAgg, TPB, 0, stream>>>(agg, N_NODES * HID);
    scatter_h_kernel<<<gSc16, TPB, 0, stream>>>(hB, src, dst, agg, N_EDGES);
    sage_wmma_kernel<<<gGemm, TPB, 0, stream>>>(agg, deg, hB, w16 + 2 * 4096, w16 + 3 * 4096, b2, hA, N_TILES);

    // layer 3
    zero_f32_kernel<<<gZeroAgg, TPB, 0, stream>>>(agg, N_NODES * HID);
    scatter_h_kernel<<<gSc16, TPB, 0, stream>>>(hA, src, dst, agg, N_EDGES);
    sage_wmma_kernel<<<gGemm, TPB, 0, stream>>>(agg, deg, hA, w16 + 4 * 4096, w16 + 5 * 4096, b3, hB, N_TILES);

    // output head
    fc_kernel<<<gFc, TPB, 0, stream>>>(hB, fw, fb, out, N_NODES);
}
